// SSMLayer_53549652247061
// MI455X (gfx1250) — matrix operand order repaired and checked by reading, not compiled, not run on hardware
//
#include <hip/hip_runtime.h>
#include <hip/hip_bf16.h>

// ---------- types ----------
typedef __attribute__((ext_vector_type(16))) __bf16 v16bf;
typedef __attribute__((ext_vector_type(8)))  __bf16 v8bf;
typedef __attribute__((ext_vector_type(8)))  float  v8f;

#define D_MODEL 1024
#define SEQ_L   4096
#define BATCH   8
#define NCHUNK  32
#define CHLEN   128           // SEQ_L / NCHUNK
#define LDSP    40            // LDS row pitch in bf16 elems (32 data + 8 pad -> 80B, conflict-free)

// ---------- fp32 -> bf16 (RNE) ----------
__device__ __forceinline__ unsigned short f2bf(float x) {
  unsigned int u = __float_as_uint(x);
  unsigned int r = (u + 0x7FFFu + ((u >> 16) & 1u)) >> 16;
  return (unsigned short)r;
}

// ---------- convert kernel: n4 = n/4 float4 groups ----------
__global__ __launch_bounds__(256) void cvt_kernel(const float* __restrict__ in,
                                                  unsigned short* __restrict__ out,
                                                  int n4) {
  int i = blockIdx.x * 256 + threadIdx.x;
  if (i >= n4) return;
  float4 v = ((const float4*)in)[i];
  uint2 p;
  p.x = (unsigned int)f2bf(v.x) | ((unsigned int)f2bf(v.y) << 16);
  p.y = (unsigned int)f2bf(v.z) | ((unsigned int)f2bf(v.w) << 16);
  ((uint2*)out)[i] = p;
}

// ---------- CDNA5 async copy: global -> LDS, 16B per lane (ASYNCcnt tracked) ----------
__device__ __forceinline__ void async_ld16(unsigned lds_addr, const unsigned short* g) {
  asm volatile("global_load_async_to_lds_b128 %0, %1, off"
               :: "v"(lds_addr), "v"(g)
               : "memory");
}
__device__ __forceinline__ void wait_async0() {
  asm volatile("s_wait_asynccnt 0x0" ::: "memory");
}

// ---------- fragment loads from LDS ----------
// A-matrix 16-bit 16x32 layout: lanes 0-15 hold K 0..7 & 16..23; lanes 16-31 hold K 8..15 & 24..31
__device__ __forceinline__ v16bf load_fragA(const unsigned short* p, int koff) {
  v8bf lo = *(const v8bf*)(p + koff);        // K = koff .. koff+7
  v8bf hi = *(const v8bf*)(p + koff + 16);   // K = koff+16 .. koff+23
  v16bf r;
#pragma unroll
  for (int i = 0; i < 8; i++) { r[i] = lo[i]; r[i + 8] = hi[i]; }
  return r;
}
// B-matrix 32x16 layout: lanes 0-15 hold K 0..7 & 16..23; lanes 16-31 hold K 8..15 & 24..31
__device__ __forceinline__ v16bf load_fragB(const unsigned short* p, int koff) {
  v8bf lo = *(const v8bf*)(p + koff);
  v8bf hi = *(const v8bf*)(p + koff + 16);
  v16bf r;
#pragma unroll
  for (int i = 0; i < 8; i++) { r[i] = lo[i]; r[i + 8] = hi[i]; }
  return r;
}

// ---------- GEMM: out[M,N] (f32) = A[M,K](bf16) * W[N,K](bf16)^T ----------
// grid = (N/256, M/128), block = 256 (8 waves; wave grid 2(M) x 4(N); wave tile 64x64)
// Double-buffered LDS, K-loop unrolled x2 so buffer indices are literals.
// Tiles staged with global_load_async_to_lds_b128 (ASYNCcnt), overlapped with WMMAs.
__global__ __launch_bounds__(256) void gemm_bf16_kernel(
    const unsigned short* __restrict__ A,
    const unsigned short* __restrict__ W,
    float* __restrict__ out,
    int M, int N, int K) {
  __shared__ __align__(16) unsigned short Alds[2][128 * LDSP];   // 2 x 10 KB
  __shared__ __align__(16) unsigned short Blds[2][256 * LDSP];   // 2 x 20 KB

  const int tid  = threadIdx.x;
  const int lane = tid & 31;
  const int wave = tid >> 5;
  const int wm   = wave >> 2;   // 0..1  -> 64 M-rows per wave
  const int wn   = wave & 3;    // 0..3  -> 64 N-cols per wave
  const int ms0  = blockIdx.y * 128;
  const int ns0  = blockIdx.x * 256;
  const int rl   = lane & 15;
  const int hi16 = (lane >> 4) & 1;
  const int koffA = hi16 ? 8 : 0;
  const int koffB = hi16 ? 8 : 0;

  // staging: A tile 128x32 = 512 16B-chunks (2/thread), B tile 256x32 = 1024 chunks (4/thread)
  const int r0 = tid >> 2;      // 0..63
  const int c0 = tid & 3;       // 16B chunk within 64B row
  const unsigned short* gA[2];
  const unsigned short* gB[4];
#pragma unroll
  for (int i = 0; i < 2; i++) gA[i] = A + (size_t)(ms0 + r0 + 64 * i) * K + c0 * 8;
#pragma unroll
  for (int i = 0; i < 4; i++) gB[i] = W + (size_t)(ns0 + r0 + 64 * i) * K + c0 * 8;
  unsigned lA[2][2], lB[2][4];
#pragma unroll
  for (int s = 0; s < 2; s++) {
#pragma unroll
    for (int i = 0; i < 2; i++) lA[s][i] = (unsigned)(size_t)&Alds[s][(r0 + 64 * i) * LDSP + c0 * 8];
#pragma unroll
    for (int i = 0; i < 4; i++) lB[s][i] = (unsigned)(size_t)&Blds[s][(r0 + 64 * i) * LDSP + c0 * 8];
  }

  // frag base pointers (per-wave, constant frag offsets fold into ds immediate offsets)
  const unsigned short* pA[2] = { &Alds[0][(wm * 64 + rl) * LDSP],
                                  &Alds[1][(wm * 64 + rl) * LDSP] };
  const unsigned short* pB[2] = { &Blds[0][(wn * 64 + rl) * LDSP],
                                  &Blds[1][(wn * 64 + rl) * LDSP] };

  v8f acc[4][4] = {};

  // preload tile 0 into buffer 0
#pragma unroll
  for (int i = 0; i < 2; i++) async_ld16(lA[0][i], gA[i]);
#pragma unroll
  for (int i = 0; i < 4; i++) async_ld16(lB[0][i], gB[i]);
  wait_async0();
  __syncthreads();

#pragma unroll 1
  for (int kk = 0; kk < K; kk += 64) {
    // ---- half-step A: compute on buffer 0, prefetch kk+32 into buffer 1 ----
    {
#pragma unroll
      for (int i = 0; i < 2; i++) async_ld16(lA[1][i], gA[i] + kk + 32);
#pragma unroll
      for (int i = 0; i < 4; i++) async_ld16(lB[1][i], gB[i] + kk + 32);

      v16bf af[4], bfr[4];
#pragma unroll
      for (int f = 0; f < 4; f++) af[f]  = load_fragA(pA[0] + f * 16 * LDSP, koffA);
#pragma unroll
      for (int g = 0; g < 4; g++) bfr[g] = load_fragB(pB[0] + g * 16 * LDSP, koffB);
#pragma unroll
      for (int f = 0; f < 4; f++)
#pragma unroll
        for (int g = 0; g < 4; g++)
          acc[f][g] = __builtin_amdgcn_wmma_f32_16x16x32_bf16(
              false, af[f], false, bfr[g], (short)0, acc[f][g], false, false);

      wait_async0();
      __syncthreads();
    }
    // ---- half-step B: compute on buffer 1, prefetch kk+64 into buffer 0 ----
    {
      if (kk + 64 < K) {
#pragma unroll
        for (int i = 0; i < 2; i++) async_ld16(lA[0][i], gA[i] + kk + 64);
#pragma unroll
        for (int i = 0; i < 4; i++) async_ld16(lB[0][i], gB[i] + kk + 64);
      }

      v16bf af[4], bfr[4];
#pragma unroll
      for (int f = 0; f < 4; f++) af[f]  = load_fragA(pA[1] + f * 16 * LDSP, koffA);
#pragma unroll
      for (int g = 0; g < 4; g++) bfr[g] = load_fragB(pB[1] + g * 16 * LDSP, koffB);
#pragma unroll
      for (int f = 0; f < 4; f++)
#pragma unroll
        for (int g = 0; g < 4; g++)
          acc[f][g] = __builtin_amdgcn_wmma_f32_16x16x32_bf16(
              false, af[f], false, bfr[g], (short)0, acc[f][g], false, false);

      wait_async0();
      __syncthreads();
    }
  }

  // C/D layout: vgpr v, lanes 0-15: M=mb+v, N=rl ; lanes 16-31: M=mb+8+v, N=rl
#pragma unroll
  for (int f = 0; f < 4; f++) {
    int mb = ms0 + wm * 64 + f * 16 + hi16 * 8;
#pragma unroll
    for (int g = 0; g < 4; g++) {
      int n = ns0 + wn * 64 + g * 16 + rl;
#pragma unroll
      for (int v = 0; v < 8; v++)
        out[(size_t)(mb + v) * N + n] = acc[f][g][v];
    }
  }
}

// ---------- scan pass 1: per-chunk decay sums (zero-seeded) ----------
__global__ __launch_bounds__(256) void scan_carry_kernel(
    const float* __restrict__ ub, const float* __restrict__ log_a,
    float* __restrict__ carry) {
  int g = blockIdx.x * 256 + threadIdx.x;    // B*NCHUNK*D threads
  int d = g & (D_MODEL - 1);
  int c = (g >> 10) & (NCHUNK - 1);
  int b = g >> 15;
  float a = __expf(log_a[d]);
  const float* p = ub + ((size_t)b * SEQ_L + (size_t)c * CHLEN) * D_MODEL + d;
  float s = 0.f;
#pragma unroll 4
  for (int t = 0; t < CHLEN; t++) { s = fmaf(a, s, *p); p += D_MODEL; }
  carry[((size_t)(b * NCHUNK + c)) * D_MODEL + d] = s;
}

// ---------- scan pass 2: serial combine across chunks; carry[] becomes carry-IN ----------
__global__ __launch_bounds__(256) void scan_combine_kernel(
    const float* __restrict__ log_a, float* __restrict__ carry) {
  int g = blockIdx.x * 256 + threadIdx.x;    // B*D threads
  int d = g & (D_MODEL - 1);
  int b = g >> 10;
  float P = __expf((float)CHLEN * log_a[d]);  // a^CHLEN
  float S = 0.f;
  float* p = carry + (size_t)b * NCHUNK * D_MODEL + d;
#pragma unroll
  for (int c = 0; c < NCHUNK; c++) {
    float sc = p[(size_t)c * D_MODEL];
    p[(size_t)c * D_MODEL] = S;              // exclusive carry-in for chunk c
    S = fmaf(P, S, sc);
  }
}

// ---------- scan pass 3: seeded re-scan, emit bf16 states ----------
__global__ __launch_bounds__(256) void scan_apply_kernel(
    const float* __restrict__ ub, const float* __restrict__ log_a,
    const float* __restrict__ carry, unsigned short* __restrict__ states) {
  int g = blockIdx.x * 256 + threadIdx.x;
  int d = g & (D_MODEL - 1);
  int c = (g >> 10) & (NCHUNK - 1);
  int b = g >> 15;
  float a = __expf(log_a[d]);
  size_t base = ((size_t)b * SEQ_L + (size_t)c * CHLEN) * D_MODEL + d;
  const float* p = ub + base;
  unsigned short* q = states + base;
  float s = carry[((size_t)(b * NCHUNK + c)) * D_MODEL + d];
#pragma unroll 4
  for (int t = 0; t < CHLEN; t++) {
    s = fmaf(a, s, *p);
    *q = f2bf(s);
    p += D_MODEL; q += D_MODEL;
  }
}

// ---------- LayerNorm over D=1024; one block per row ----------
__global__ __launch_bounds__(256) void ln_kernel(
    const float* __restrict__ y, const float* __restrict__ gamma,
    const float* __restrict__ beta, float* __restrict__ out) {
  __shared__ float sh1[256];
  __shared__ float sh2[256];
  int row = blockIdx.x;
  int t = threadIdx.x;
  const float* yr = y + (size_t)row * D_MODEL;
  float4 v = *(const float4*)(yr + t * 4);
  float s  = v.x + v.y + v.z + v.w;
  float s2 = v.x * v.x + v.y * v.y + v.z * v.z + v.w * v.w;
  sh1[t] = s; sh2[t] = s2;
  __syncthreads();
#pragma unroll
  for (int o = 128; o > 0; o >>= 1) {
    if (t < o) { sh1[t] += sh1[t + o]; sh2[t] += sh2[t + o]; }
    __syncthreads();
  }
  float mean = sh1[0] * (1.f / D_MODEL);
  float var  = sh2[0] * (1.f / D_MODEL) - mean * mean;
  float inv  = rsqrtf(var + 1e-5f);
  float4 gv = *(const float4*)(gamma + t * 4);
  float4 bv = *(const float4*)(beta + t * 4);
  float4 o4;
  o4.x = (v.x - mean) * inv * gv.x + bv.x;
  o4.y = (v.y - mean) * inv * gv.y + bv.y;
  o4.z = (v.z - mean) * inv * gv.z + bv.z;
  o4.w = (v.w - mean) * inv * gv.w + bv.w;
  *(float4*)(out + (size_t)row * D_MODEL + t * 4) = o4;
}

// ---------- launch ----------
extern "C" void kernel_launch(void* const* d_in, const int* in_sizes, int n_in,
                              void* d_out, int out_size, void* d_ws, size_t ws_size,
                              hipStream_t stream) {
  (void)in_sizes; (void)n_in; (void)out_size; (void)ws_size;
  const float* u     = (const float*)d_in[0];
  const float* log_a = (const float*)d_in[1];
  const float* W_b   = (const float*)d_in[2];
  const float* W_c   = (const float*)d_in[3];
  const float* gamma = (const float*)d_in[4];
  const float* beta  = (const float*)d_in[5];
  float* out = (float*)d_out;

  const int M = BATCH * SEQ_L;      // 32768
  const int D = D_MODEL;            // 1024
  const size_t NU = (size_t)M * D;  // 33,554,432

  char* ws = (char*)d_ws;
  unsigned short* u_bf  = (unsigned short*)ws;                              // 64 MB
  unsigned short* wb_bf = (unsigned short*)(ws + NU * 2);                   //  2 MB
  unsigned short* wc_bf = (unsigned short*)(ws + NU * 2 + (size_t)D * D * 2);
  float*          ubuf  = (float*)(ws + NU * 2 + (size_t)D * D * 4);        // 128 MB
  unsigned short* st_bf = (unsigned short*)((char*)ubuf + NU * 4);          // 64 MB
  float*          carry = (float*)((char*)st_bf + NU * 2);                  //  1 MB
  float*          ybuf  = ubuf;  // reuse: ub dead after scan pass 3

  // 1) fp32 -> bf16
  cvt_kernel<<<(unsigned)(NU / 4 / 256), 256, 0, stream>>>(u, u_bf, (int)(NU / 4));
  cvt_kernel<<<(D * D / 4) / 256, 256, 0, stream>>>(W_b, wb_bf, D * D / 4);
  cvt_kernel<<<(D * D / 4) / 256, 256, 0, stream>>>(W_c, wc_bf, D * D / 4);

  // 2) ub = u @ W_b^T
  dim3 gg(D / 256, M / 128);
  gemm_bf16_kernel<<<gg, 256, 0, stream>>>(u_bf, wb_bf, ubuf, M, D, D);

  // 3) parallel scan over L
  scan_carry_kernel<<<(BATCH * NCHUNK * D_MODEL) / 256, 256, 0, stream>>>(ubuf, log_a, carry);
  scan_combine_kernel<<<(BATCH * D_MODEL) / 256, 256, 0, stream>>>(log_a, carry);
  scan_apply_kernel<<<(BATCH * NCHUNK * D_MODEL) / 256, 256, 0, stream>>>(ubuf, log_a, carry, st_bf);

  // 4) y = states @ W_c^T  (writes into ub's space)
  gemm_bf16_kernel<<<gg, 256, 0, stream>>>(st_bf, wc_bf, ybuf, M, D, D);

  // 5) LayerNorm
  ln_kernel<<<M, 256, 0, stream>>>(ybuf, gamma, beta, out);
}
